// SparseTransformerBlock_90838558310634
// MI455X (gfx1250) — matrix operand order repaired and checked by reading, not compiled, not run on hardware
//
#include <hip/hip_runtime.h>
#include <hip/hip_bf16.h>

typedef __attribute__((ext_vector_type(16))) _Float16 v16h;
typedef __attribute__((ext_vector_type(8)))  _Float16 v8h;
typedef __attribute__((ext_vector_type(8)))  float    v8f;
typedef __attribute__((ext_vector_type(4)))  int      v4i;

union V16 { v16h v; v8h h[2]; _Float16 f[16]; };

// Problem constants
constexpr int Bc = 2, Sc = 2048, Ec = 1024, Hc = 16, MLPc = 4096, Dc = 64;
constexpr int Mc = Bc * Sc;          // 4096 rows
constexpr float SCALE = 0.125f;      // 1/sqrt(64)

// ---------------------------------------------------------------------------
// CDNA5 async global->LDS copy (ASYNCcnt-tracked) with fallback to inline asm
// Builtin signature (from hipcc diagnostic): (v4i __device__*, v4i __shared__*,
// imm offset, imm cpol)
// ---------------------------------------------------------------------------
typedef __attribute__((address_space(1))) v4i glb_v4i;
typedef __attribute__((address_space(3))) v4i lds_v4i;

static __device__ __forceinline__ void g2l_async16(const void* g, void* l) {
#if __has_builtin(__builtin_amdgcn_global_load_async_to_lds_b128)
    __builtin_amdgcn_global_load_async_to_lds_b128(
        (glb_v4i*)(unsigned long long)g,
        (lds_v4i*)(unsigned int)(unsigned long long)l,
        0, 0);
#else
    unsigned int       la = (unsigned int)(unsigned long long)l;  // LDS offset
    unsigned long long ga = (unsigned long long)g;
    asm volatile("global_load_async_to_lds_b128 %0, %1, off"
                 :: "v"(la), "v"(ga) : "memory");
#endif
}

static __device__ __forceinline__ void wait_async_le4() {
#if __has_builtin(__builtin_amdgcn_s_wait_asynccnt)
    __builtin_amdgcn_s_wait_asynccnt(4);
#else
    asm volatile("s_wait_asynccnt 0x4" ::: "memory");
#endif
}
static __device__ __forceinline__ void wait_async_le0() {
#if __has_builtin(__builtin_amdgcn_s_wait_asynccnt)
    __builtin_amdgcn_s_wait_asynccnt(0);
#else
    asm volatile("s_wait_asynccnt 0x0" ::: "memory");
#endif
}

// ---------------------------------------------------------------------------
// f32 -> f16 conversion (weights)
// ---------------------------------------------------------------------------
__global__ void cvt_f32_to_f16(const float* __restrict__ in,
                               _Float16* __restrict__ out, int n) {
    int i = blockIdx.x * blockDim.x + threadIdx.x;
    if (i < n) out[i] = (_Float16)in[i];
}

// ---------------------------------------------------------------------------
// LayerNorm over last dim (E) -> f16 output. One 256-thread block per row.
// ---------------------------------------------------------------------------
__global__ void ln_to_f16(const float* __restrict__ in,
                          const float* __restrict__ g,
                          const float* __restrict__ bta,
                          _Float16* __restrict__ out, int E) {
    const int row = blockIdx.x;
    const int tid = threadIdx.x;
    const float* p = in + (size_t)row * E;
    __shared__ float red[256];

    float s = 0.f;
    for (int i = tid; i < E; i += 256) s += p[i];
    red[tid] = s; __syncthreads();
    for (int st = 128; st > 0; st >>= 1) {
        if (tid < st) red[tid] += red[tid + st];
        __syncthreads();
    }
    const float mu = red[0] / (float)E;
    __syncthreads();

    float v = 0.f;
    for (int i = tid; i < E; i += 256) { float d = p[i] - mu; v += d * d; }
    red[tid] = v; __syncthreads();
    for (int st = 128; st > 0; st >>= 1) {
        if (tid < st) red[tid] += red[tid + st];
        __syncthreads();
    }
    const float rs = rsqrtf(red[0] / (float)E + 1e-5f);

    _Float16* o = out + (size_t)row * E;
    for (int i = tid; i < E; i += 256)
        o[i] = (_Float16)((p[i] - mu) * rs * g[i] + bta[i]);
}

// ---------------------------------------------------------------------------
// WMMA GEMM: C[M,N] = A[M,K] * W[N,K]^T (+bias[N]) (+resid[M,N])
// Block (256 thr / 8 waves) -> 128x128 C tile. K consumed in 32-wide slabs
// staged into double-buffered LDS via async global->LDS B128 copies
// (ASYNCcnt), so WMMAs on slab k overlap the DMA of slab k+1.
// Wave tile: 32x64 (2 M-tiles x 4 N-tiles). Fragment layouts used here:
//   A (16x32): lane holds row m=lane%16, k = e + (lane/16)*16 (contiguous)
//   B (32x16): lane holds col n=lane%16, k = e + (lane/16)*16 (contiguous)
//   C (16x16): vgpr r -> row r + (lane/16)*8, col = lane%16
// ---------------------------------------------------------------------------
__global__ __launch_bounds__(256)
void gemm_wmma(const _Float16* __restrict__ A,
               const _Float16* __restrict__ W,
               const float* __restrict__ bias,
               const float* __restrict__ resid,
               float* __restrict__ outF,
               _Float16* __restrict__ outH,
               int Mb128, int Nb128, int K) {
    __shared__ __align__(16) _Float16 bufA[2][128 * 32];
    __shared__ __align__(16) _Float16 bufB[2][128 * 32];

    const int tid    = threadIdx.x;
    const int wid    = tid >> 5;
    const int lane   = tid & 31;
    const int lane16 = lane & 15;
    const int half   = lane >> 4;

    const int bm   = blockIdx.x / Nb128;
    const int bn   = blockIdx.x % Nb128;
    const int Mblk = bm * 128;
    const int Nblk = bn * 128;
    const int moff = (wid >> 1) * 32;     // wave's M offset inside block tile
    const int noff = (wid & 1) * 64;      // wave's N offset inside block tile

    v8f acc[2][4];
#pragma unroll
    for (int mt = 0; mt < 2; ++mt)
#pragma unroll
        for (int j = 0; j < 4; ++j) acc[mt][j] = (v8f)0.0f;

    const int nk = K / 32;

    // Stage one 128x32 slab of A and of W into LDS buffer `buf`.
    // 512 b128 units per panel / 256 threads = 2 per thread per panel
    // -> 4 async issues per thread per slab.
    auto issue = [&](int kb, int buf) {
#pragma unroll
        for (int r = 0; r < 2; ++r) {
            const int u   = tid + 256 * r;
            const int row = u >> 1;
            const int col = (u & 1) * 16;
            g2l_async16(A + (size_t)(Mblk + row) * K + kb * 32 + col,
                        &bufA[buf][row * 32 + col]);
            g2l_async16(W + (size_t)(Nblk + row) * K + kb * 32 + col,
                        &bufB[buf][row * 32 + col]);
        }
    };

    issue(0, 0);
    for (int kb = 0; kb < nk; ++kb) {
        const int cur = kb & 1;
        if (kb + 1 < nk) {
            issue(kb + 1, cur ^ 1);               // DMA next slab
            if (kb + 2 < nk) {                    // warm L2 two slabs ahead
                const int row = tid >> 1, col = (tid & 1) * 16;
                __builtin_prefetch(A + (size_t)(Mblk + row) * K + (kb + 2) * 32 + col, 0, 0);
                __builtin_prefetch(W + (size_t)(Nblk + row) * K + (kb + 2) * 32 + col, 0, 0);
            }
            wait_async_le4();                     // slab kb complete, kb+1 in flight
        } else {
            wait_async_le0();
        }
        __syncthreads();

        V16 a[2];
#pragma unroll
        for (int mt = 0; mt < 2; ++mt) {
            const _Float16* ar = &bufA[cur][(moff + mt * 16 + lane16) * 32];
            a[mt].h[0] = *(const v8h*)(ar + half * 16);
            a[mt].h[1] = *(const v8h*)(ar + half * 16 + 8);
        }
#pragma unroll
        for (int j = 0; j < 4; ++j) {
            const _Float16* br = &bufB[cur][(noff + j * 16 + lane16) * 32];
            V16 bw;
            bw.h[0] = *(const v8h*)(br + half * 16);
            bw.h[1] = *(const v8h*)(br + half * 16 + 8);
#pragma unroll
            for (int mt = 0; mt < 2; ++mt)
                acc[mt][j] = __builtin_amdgcn_wmma_f32_16x16x32_f16(
                    false, a[mt].v, false, bw.v, (short)0, acc[mt][j], false, false);
        }
        __syncthreads();   // all waves done with `cur` before it is re-filled
    }

    const int N = Nb128 * 128;
#pragma unroll
    for (int mt = 0; mt < 2; ++mt)
#pragma unroll
        for (int j = 0; j < 4; ++j) {
            const int col = Nblk + noff + j * 16 + lane16;
#pragma unroll
            for (int r = 0; r < 8; ++r) {
                const int row = Mblk + moff + mt * 16 + r + half * 8;
                float v = acc[mt][j][r];
                if (bias)  v += bias[col];
                if (resid) v += resid[(size_t)row * N + col];
                if (outF)  outF[(size_t)row * N + col] = v;
                if (outH)  outH[(size_t)row * N + col] = (_Float16)v;
            }
        }
}

// ---------------------------------------------------------------------------
// Flash-style masked attention. One wave per (b, h, 16-query tile).
// qkv: f16 [B*S, 3E] rows = {q | k | v}, head-major inside each E chunk.
// ctx: f16 [B*S, E].
// ---------------------------------------------------------------------------
__global__ void attn_flash(const _Float16* __restrict__ qkv,
                           const unsigned char* __restrict__ mask,
                           _Float16* __restrict__ ctx) {
    const int lane   = threadIdx.x;
    const int lane16 = lane & 15;
    const int half   = lane >> 4;

    const int bid = blockIdx.x;
    const int qt  = bid & (Sc / 16 - 1);        // 128 query tiles
    const int h   = (bid >> 7) & (Hc - 1);
    const int b   = bid >> 11;

    __shared__ __align__(16) float    s_sc[16][32];
    __shared__ __align__(16) _Float16 s_p[16][32];
    __shared__ __align__(16) _Float16 s_v[32][64];
    __shared__ float s_m[16], s_l[16], s_c[16];

    // Q fragment (A layout), loaded once: two K=32 chunks over D=64
    const size_t qrow = (size_t)(b * Sc + qt * 16 + lane16) * (3 * Ec) + h * Dc;
    V16 qa[2];
#pragma unroll
    for (int c = 0; c < 2; ++c) {
        qa[c].h[0] = *(const v8h*)(qkv + qrow + c * 32 + half * 16);
        qa[c].h[1] = *(const v8h*)(qkv + qrow + c * 32 + half * 16 + 8);
    }

    v8f acc[4];
#pragma unroll
    for (int c = 0; c < 4; ++c) acc[c] = (v8f)0.0f;

    if (lane < 16) { s_m[lane] = -INFINITY; s_l[lane] = 0.f; }
    __syncthreads();

    for (int k0 = 0; k0 < Sc; k0 += 32) {
        // ---- stage V tile [32 keys][64 d] into LDS (coalesced 16B loads) ----
#pragma unroll
        for (int t = 0; t < 8; ++t) {
            const int u  = lane + 32 * t;       // 256 v8h units
            const int kl = u >> 3;
            const int dc = u & 7;
            *(v8h*)(&s_v[kl][dc * 8]) =
                *(const v8h*)(qkv + (size_t)(b * Sc + k0 + kl) * (3 * Ec)
                              + 2 * Ec + h * Dc + dc * 8);
        }

        // ---- scores: two 16x16 tiles, K-contraction over D=64 (2 WMMAs each)
#pragma unroll
        for (int t = 0; t < 2; ++t) {
            v8f sc = (v8f)0.0f;
            const int key = k0 + t * 16 + lane16;
            const size_t krow = (size_t)(b * Sc + key) * (3 * Ec) + Ec + h * Dc;
#pragma unroll
            for (int c = 0; c < 2; ++c) {
                V16 bk;
                bk.h[0] = *(const v8h*)(qkv + krow + c * 32 + half * 16);
                bk.h[1] = *(const v8h*)(qkv + krow + c * 32 + half * 16 + 8);
                sc = __builtin_amdgcn_wmma_f32_16x16x32_f16(
                    false, qa[c].v, false, bk.v, (short)0, sc, false, false);
            }
#pragma unroll
            for (int r = 0; r < 8; ++r) {
                const int q = qt * 16 + r + half * 8;
                float v = sc[r] * SCALE;
                v = mask[(size_t)q * Sc + key] ? v : -1e9f;
                s_sc[r + half * 8][t * 16 + lane16] = v;
            }
        }
        __syncthreads();

        // ---- online softmax state update (one lane per query row) ----
        if (lane < 16) {
            float mo = s_m[lane], mm = mo;
            for (int c = 0; c < 32; ++c) mm = fmaxf(mm, s_sc[lane][c]);
            const float corr = __expf(mo - mm);
            float l = s_l[lane] * corr;
            for (int c = 0; c < 32; ++c) {
                const float p = __expf(s_sc[lane][c] - mm);
                l += p;
                s_p[lane][c] = (_Float16)p;
            }
            s_m[lane] = mm; s_l[lane] = l; s_c[lane] = corr;
        }
        __syncthreads();

        // ---- rescale context accumulators by per-row correction ----
        float corrv[8];
#pragma unroll
        for (int r = 0; r < 8; ++r) corrv[r] = s_c[r + half * 8];
#pragma unroll
        for (int c = 0; c < 4; ++c)
#pragma unroll
            for (int r = 0; r < 8; ++r) acc[c][r] *= corrv[r];

        // ---- P fragment (A layout) from LDS: contiguous 16B chunks ----
        V16 pa;
        pa.h[0] = *(const v8h*)(&s_p[lane16][half * 16]);
        pa.h[1] = *(const v8h*)(&s_p[lane16][half * 16 + 8]);

        // ---- ctx += P(16x32) x V(32x16) for 4 d-chunks ----
#pragma unroll
        for (int c = 0; c < 4; ++c) {
            V16 bv;
#pragma unroll
            for (int e = 0; e < 16; ++e)
                bv.f[e] = s_v[half * 16 + e][c * 16 + lane16];
            acc[c] = __builtin_amdgcn_wmma_f32_16x16x32_f16(
                false, pa.v, false, bv.v, (short)0, acc[c], false, false);
        }
        __syncthreads();
    }

    // ---- normalize and store context (f16) ----
    float lv[8];
#pragma unroll
    for (int r = 0; r < 8; ++r) lv[r] = s_l[r + half * 8];
#pragma unroll
    for (int c = 0; c < 4; ++c)
#pragma unroll
        for (int r = 0; r < 8; ++r) {
            const int q = qt * 16 + r + half * 8;
            ctx[(size_t)(b * Sc + q) * Ec + h * Dc + c * 16 + lane16] =
                (_Float16)(acc[c][r] / lv[r]);
        }
}

// ---------------------------------------------------------------------------
extern "C" void kernel_launch(void* const* d_in, const int* in_sizes, int n_in,
                              void* d_out, int out_size, void* d_ws, size_t ws_size,
                              hipStream_t stream) {
    const float*         x     = (const float*)d_in[0];
    const unsigned char* mask  = (const unsigned char*)d_in[1];
    const float*         w_qkv = (const float*)d_in[2];
    const float*         b_qkv = (const float*)d_in[3];
    const float*         w_out = (const float*)d_in[4];
    const float*         b_out = (const float*)d_in[5];
    const float*         g1    = (const float*)d_in[6];
    const float*         beta1 = (const float*)d_in[7];
    const float*         g2    = (const float*)d_in[8];
    const float*         beta2 = (const float*)d_in[9];
    const float*         w1    = (const float*)d_in[10];
    const float*         bias1 = (const float*)d_in[11];
    const float*         w2    = (const float*)d_in[12];
    const float*         bias2 = (const float*)d_in[13];
    float*               out   = (float*)d_out;

    char* wsp = (char*)d_ws;
    auto alloc = [&](size_t bytes) -> char* {
        char* p = wsp;
        wsp += (bytes + 255) & ~(size_t)255;
        return p;
    };
    _Float16* wqkv16 = (_Float16*)alloc((size_t)3 * Ec * Ec * 2);
    _Float16* wout16 = (_Float16*)alloc((size_t)Ec * Ec * 2);
    _Float16* w1_16  = (_Float16*)alloc((size_t)MLPc * Ec * 2);
    _Float16* w2_16  = (_Float16*)alloc((size_t)Ec * MLPc * 2);
    _Float16* h16    = (_Float16*)alloc((size_t)Mc * Ec * 2);
    _Float16* qkv16  = (_Float16*)alloc((size_t)Mc * 3 * Ec * 2);
    _Float16* ctx16  = (_Float16*)alloc((size_t)Mc * Ec * 2);
    float*    attnrs = (float*)   alloc((size_t)Mc * Ec * 4);
    _Float16* h2_16  = (_Float16*)alloc((size_t)Mc * Ec * 2);
    _Float16* mlp16  = (_Float16*)alloc((size_t)Mc * MLPc * 2);

    auto cvt = [&](const float* src, _Float16* dst, int n) {
        cvt_f32_to_f16<<<(n + 255) / 256, 256, 0, stream>>>(src, dst, n);
    };
    cvt(w_qkv, wqkv16, 3 * Ec * Ec);
    cvt(w_out, wout16, Ec * Ec);
    cvt(w1,    w1_16,  MLPc * Ec);
    cvt(w2,    w2_16,  Ec * MLPc);

    auto gemm = [&](const _Float16* A, const _Float16* W, const float* bias,
                    const float* resid, float* oF, _Float16* oH,
                    int M, int N, int K) {
        const int blocks = (M / 128) * (N / 128);
        gemm_wmma<<<blocks, 256, 0, stream>>>(
            A, W, bias, resid, oF, oH, M / 128, N / 128, K);
    };

    // h = LN1(x)
    ln_to_f16<<<Mc, 256, 0, stream>>>(x, g1, beta1, h16, Ec);
    // qkv = h @ w_qkv^T + b_qkv
    gemm(h16, wqkv16, b_qkv, nullptr, nullptr, qkv16, Mc, 3 * Ec, Ec);
    // masked flash attention -> ctx
    attn_flash<<<Bc * Hc * (Sc / 16), 32, 0, stream>>>(qkv16, mask, ctx16);
    // attn_out = ctx @ w_out^T + b_out + x   (residual 1)
    gemm(ctx16, wout16, b_out, x, attnrs, nullptr, Mc, Ec, Ec);
    // h2 = LN2(attn_out)
    ln_to_f16<<<Mc, 256, 0, stream>>>(attnrs, g2, beta2, h2_16, Ec);
    // m = h2 @ w1^T + bias1
    gemm(h2_16, w1_16, bias1, nullptr, nullptr, mlp16, Mc, MLPc, Ec);
    // out = m @ w2^T + bias2 + x   (residual adds ORIGINAL x)
    gemm(mlp16, w2_16, bias2, x, out, nullptr, Mc, Ec, MLPc);
}
